// QuantLinear_36936718745876
// MI455X (gfx1250) — compile-verified, matrix operand order checked by reading, not run on hardware
//
#include <hip/hip_runtime.h>
#include <hip/hip_bf16.h>

typedef _Float16 v4h  __attribute__((ext_vector_type(4)));
typedef _Float16 v8h  __attribute__((ext_vector_type(8)));
typedef _Float16 v16h __attribute__((ext_vector_type(16)));
typedef float    v8f  __attribute__((ext_vector_type(8)));

#define IN_F   4096
#define OUT_F  11008
#define GROUP  128

#define BM 128
#define BN 128
#define BK 64
#define LDK 80          // padded K-stride in halves (64 + 16): 160B rows, 16B-aligned ds ops
#define THREADS 256
#define KT (IN_F / BK)  // 64 K-iterations
#define LDS_TILE ((BM + BN) * LDK)

// Concatenate two 16-byte LDS fragments into a v16h WMMA operand.
static __device__ inline v16h frag_ld(const _Float16* p) {
    v8h lo = *(const v8h*)(p);
    v8h hi = *(const v8h*)(p + 16);
    return __builtin_shufflevector(lo, hi, 0,1,2,3,4,5,6,7,8,9,10,11,12,13,14,15);
}

__global__ __launch_bounds__(THREADS, 2)
void wqlinear_wmma_kernel(const float* __restrict__ x,
                          const int*   __restrict__ qweight,
                          const int*   __restrict__ qzeros,
                          const float* __restrict__ scales,
                          const float* __restrict__ bias,
                          float*       __restrict__ out,
                          int M)
{
    // Double-buffered: per buffer As [BM][LDK] f16, Bs [BN][LDK] f16 (k-major)
    __shared__ _Float16 lds[2][LDS_TILE];

    const int tid  = threadIdx.x;
    const int lane = tid & 31;
    const int wave = tid >> 5;
    const int l15  = lane & 15;
    const int lhi  = lane >> 4;

    const int tileN = blockIdx.x * BN;
    const int tileM = blockIdx.y * BM;

    // 8 waves: 4 along M (32 rows each) x 2 along N (64 cols each)
    const int wm = (wave & 3) * 32;
    const int wn = (wave >> 2) * 64;

    v8f acc[2][4];
    #pragma unroll
    for (int mi = 0; mi < 2; ++mi)
        #pragma unroll
        for (int ni = 0; ni < 4; ++ni)
            acc[mi][ni] = (v8f)(0.0f);

    // --- global-load thread mapping ---
    const int xf  = tid & 15;   // float4 index within an x row (16 * 4 = 64 k)
    const int xr0 = tid >> 4;   // x row group base (16 rows/group, 8 groups)
    const int qn  = tid & 127;  // weight column within tile
    const int qk  = tid >> 7;   // packed-word row parity (0..1)

    float4 xreg[8];
    int    wreg[4];
    float  wscale, wzs;

    const float* xrow[8];
    #pragma unroll
    for (int i = 0; i < 8; ++i) {
        int grow = tileM + xr0 + 16 * i;
        grow = grow < M ? grow : (M - 1);      // clamp: keep EXEC all-ones
        xrow[i] = x + (size_t)grow * IN_F + xf * 4;
    }
    const int* qcol = qweight + tileN + qn;    // [packed_row][col] walk

    auto load_tile = [&](int kt) {
        const int kbase = kt * BK;
        #pragma unroll
        for (int i = 0; i < 8; ++i)
            xreg[i] = *(const float4*)(xrow[i] + kbase);

        const int kw0 = kt * (BK / 8);         // qweight packed-row base
        #pragma unroll
        for (int i = 0; i < 4; ++i)
            wreg[i] = qcol[(size_t)(kw0 + qk + 2 * i) * OUT_F];

        // one quant group per BK=64 tile (64 | 128)
        const int g = kbase / GROUP;
        const float s  = scales[(size_t)g * OUT_F + tileN + qn];
        const int  zw  = qzeros[(size_t)g * (OUT_F / 8) + ((tileN + qn) >> 3)];
        const int  z   = (zw >> ((qn & 7) * 4)) & 15;
        wscale = s;
        wzs    = -s * (float)z;                // w = q*s + (-z*s)

        // L2 prefetch one more tile ahead (speculative; dropped if unmapped)
        const int pk = (kt + 1 < KT) ? (kt + 1) : kt;
        __builtin_prefetch(xrow[0] + pk * BK, 0, 1);
        __builtin_prefetch(qcol + (size_t)(pk * (BK / 8) + qk) * OUT_F, 0, 1);
    };

    auto store_tile = [&](int buf) {
        _Float16* As = lds[buf];
        _Float16* Bs = lds[buf] + BM * LDK;
        #pragma unroll
        for (int i = 0; i < 8; ++i) {
            v4h h;
            h.x = (_Float16)xreg[i].x;  h.y = (_Float16)xreg[i].y;
            h.z = (_Float16)xreg[i].z;  h.w = (_Float16)xreg[i].w;
            *(v4h*)(As + (xr0 + 16 * i) * LDK + xf * 4) = h;   // 8B ds_store
        }
        #pragma unroll
        for (int i = 0; i < 4; ++i) {
            const int word = wreg[i];
            v8h wv;
            #pragma unroll
            for (int j = 0; j < 8; ++j) {
                int q = (word >> (4 * j)) & 15;
                wv[j] = (_Float16)fmaf((float)q, wscale, wzs);
            }
            *(v8h*)(Bs + qn * LDK + (qk + 2 * i) * 8) = wv;    // 16B ds_store
        }
    };

    auto compute = [&](int buf) {
        const _Float16* As = lds[buf];
        const _Float16* Bs = lds[buf] + BM * LDK;
        #pragma unroll
        for (int ks = 0; ks < BK; ks += 32) {
            v16h afrag[2], bfrag[4];
            #pragma unroll
            for (int mi = 0; mi < 2; ++mi)
                afrag[mi] = frag_ld(As + (wm + mi * 16 + l15) * LDK + ks + lhi * 8);
            #pragma unroll
            for (int ni = 0; ni < 4; ++ni)
                bfrag[ni] = frag_ld(Bs + (wn + ni * 16 + l15) * LDK + ks + lhi * 8);
            #pragma unroll
            for (int mi = 0; mi < 2; ++mi)
                #pragma unroll
                for (int ni = 0; ni < 4; ++ni)
                    acc[mi][ni] = __builtin_amdgcn_wmma_f32_16x16x32_f16(
                        false, afrag[mi], false, bfrag[ni],
                        (short)0, acc[mi][ni], false, false);
        }
    };

    // --- 3-stage pipeline: regs(kt+2) / LDS-other-buffer(kt+1) / compute(kt) ---
    load_tile(0);          // tile 0 -> regs
    store_tile(0);         // tile 0 -> LDS[0]
    load_tile(1);          // tile 1 -> regs
    __syncthreads();       // LDS[0] ready

    for (int kt = 0; kt < KT; ++kt) {
        const int buf = kt & 1;
        if (kt + 1 < KT) store_tile(buf ^ 1);   // stage kt+1 while computing kt
        if (kt + 2 < KT) load_tile(kt + 2);     // issue global loads for kt+2
        compute(buf);
        __syncthreads();   // kt+1 buffer complete; reads of buf retired
    }

    // --- epilogue: bias add + coalesced fp32 stores ---
    #pragma unroll
    for (int ni = 0; ni < 4; ++ni) {
        const int n = tileN + wn + ni * 16 + l15;
        const float b = bias[n];
        #pragma unroll
        for (int mi = 0; mi < 2; ++mi) {
            const int mbase = tileM + wm + mi * 16 + lhi * 8;
            #pragma unroll
            for (int r = 0; r < 8; ++r) {
                const int m = mbase + r;
                if (m < M)
                    out[(size_t)m * OUT_F + n] = acc[mi][ni][r] + b;
            }
        }
    }
}

extern "C" void kernel_launch(void* const* d_in, const int* in_sizes, int n_in,
                              void* d_out, int out_size, void* d_ws, size_t ws_size,
                              hipStream_t stream) {
    const float* x       = (const float*)d_in[0];
    const int*   qweight = (const int*)  d_in[1];
    const int*   qzeros  = (const int*)  d_in[2];
    const float* scales  = (const float*)d_in[3];
    const float* bias    = (const float*)d_in[4];
    float*       out     = (float*)d_out;

    const int M = in_sizes[0] / IN_F;   // 2*2048 = 4096 rows
    dim3 grid(OUT_F / BN, (M + BM - 1) / BM);
    wqlinear_wmma_kernel<<<grid, THREADS, 0, stream>>>(
        x, qweight, qzeros, scales, bias, out, M);
}
